// CAMixer0_18820546691540
// MI455X (gfx1250) — matrix operand order repaired and checked by reading, not compiled, not run on hardware
//
#include <hip/hip_runtime.h>

typedef __attribute__((ext_vector_type(16))) __bf16 v16bf;
typedef __attribute__((ext_vector_type(8)))  __bf16 v8bf;
typedef __attribute__((ext_vector_type(8)))  float  v8f;
typedef __attribute__((ext_vector_type(4)))  unsigned int v4u;
typedef __attribute__((ext_vector_type(8)))  int v8i;
typedef __attribute__((ext_vector_type(4)))  int v4i;

#define CD  96
#define HH  192
#define WWD 192
#define HW  (HH * WWD)
#define WINS_PER_BLOCK 8
#define NBLK 576            // 4608 windows / 8

// ---------------- LDS layout (byte offsets; dynamic LDS starts at 0) -------
#define OFF_W     0          // bf16 [Wq|Wk|Wv]           55296
#define OFF_S0    55296      // f32 x-tile stage buf 0    24576
#define OFF_S1    79872      // f32 x-tile stage buf 1    24576
#define OFF_SX    104448     // bf16 x tile [64][96]      12288
#define OFF_SQ    116736     // bf16 q [64][96]           12288
#define OFF_SK    129024     // bf16 k [64][96]           12288
#define OFF_SVT   141312     // bf16 v^T [96][64]         12288
#define OFF_SAG   153600     // bf16 agent [16][96]        3072
#define OFF_SA1   156672     // bf16 attn1 [16][64]        2048
#define OFF_SAVT  158720     // bf16 agent_v^T [96][32]    6144
#define OFF_SA2   164864     // bf16 attn2 [64][32]        4096
#define OFF_SL1   168960     // f32 logits1 [16][64]       4096
#define OFF_SL2   173056     // f32 logits2 [64][16]       4096
#define OFF_SFT   177152     // f32 f^T [96][64]          24576
#define SMEM_BYTES 201728

#ifndef __has_builtin
#define __has_builtin(x) 0
#endif
#if __has_builtin(__builtin_amdgcn_tensor_load_to_lds)
#define HAVE_TDM 1
#else
#define HAVE_TDM 0
#endif
#if defined(__has_include)
#if __has_include(<hip/amd_detail/amd_gfx1250_TDM.h>)
#define TDM_6ARG 1
#endif
#endif

__device__ __forceinline__ __bf16 f2bf(float f) {
  union { float f; unsigned u; } a; a.f = f;
  unsigned r = (a.u + 0x7FFFu + ((a.u >> 16) & 1u)) >> 16;
  union { unsigned short s; __bf16 b; } o; o.s = (unsigned short)r;
  return o.b;
}
__device__ __forceinline__ float bf2f(__bf16 b) {
  union { unsigned short s; __bf16 b; } i; i.b = b;
  union { unsigned u; float f; } o; o.u = ((unsigned)i.s) << 16;
  return o.f;
}
__device__ __forceinline__ v8f vzero() {
  v8f z = {0.f, 0.f, 0.f, 0.f, 0.f, 0.f, 0.f, 0.f};
  return z;
}

// ---------------- TDM descriptor helpers (D# per CDNA5 ISA ch.8) ----------
#if HAVE_TDM
__device__ __forceinline__ v4u tdm_g0(unsigned lds_addr, unsigned long long gaddr) {
  v4u g;
  g[0] = 1u;                                              // count=1, user mode
  g[1] = lds_addr;                                        // LDS byte address
  g[2] = (unsigned)(gaddr & 0xFFFFFFFFull);               // global_addr[31:0]
  g[3] = (unsigned)((gaddr >> 32) & 0x1FFFFFFull) | (2u << 30); // [56:32] | type=2
  return g;
}
// dims/strides in data_size units. tile_dim1/2 = 0 -> unused (lower-D walk).
__device__ __forceinline__ v8i tdm_g1(int ds_code, unsigned d0, unsigned d1,
                                      unsigned t0, unsigned t1, unsigned t2,
                                      unsigned long long s0, unsigned long long s1) {
  v8i g;
  g[0] = (int)((unsigned)ds_code << 16);                  // wg_mask=0, data_size
  g[1] = (int)((d0 & 0xFFFFu) << 16);                     // tensor_dim0[15:0]
  g[2] = (int)((d0 >> 16) | ((d1 & 0xFFFFu) << 16));      // dim0[31:16] | dim1[15:0]
  g[3] = (int)((d1 >> 16) | (t0 << 16));                  // dim1[31:16] | tile_dim0
  g[4] = (int)(t1 | (t2 << 16));                          // tile_dim1 | tile_dim2
  g[5] = (int)(s0 & 0xFFFFFFFFull);                       // dim0_stride[31:0]
  g[6] = (int)(((s0 >> 32) & 0xFFFFull) | ((s1 & 0xFFFFull) << 16));
  g[7] = (int)((s1 >> 16) & 0xFFFFFFFFull);               // dim1_stride[47:16]
  return g;
}
__device__ __forceinline__ v4i tdm_g2(unsigned d2) {
  v4i g; g[0] = (int)d2; g[1] = 0; g[2] = 0; g[3] = 0; return g;
}
__device__ __forceinline__ void tdm_load(v4u g0, v8i g1, v4i g2, v4i g3) {
#ifdef TDM_6ARG
  v8i z8 = {0, 0, 0, 0, 0, 0, 0, 0};
  __builtin_amdgcn_tensor_load_to_lds(g0, g1, g2, g3, z8, 0);
#else
  __builtin_amdgcn_tensor_load_to_lds(g0, g1, g2, g3, 0);
#endif
}
__device__ __forceinline__ void tdm_store(v4u g0, v8i g1, v4i g2, v4i g3) {
#ifdef TDM_6ARG
  v8i z8 = {0, 0, 0, 0, 0, 0, 0, 0};
  __builtin_amdgcn_tensor_store_from_lds(g0, g1, g2, g3, z8, 0);
#else
  __builtin_amdgcn_tensor_store_from_lds(g0, g1, g2, g3, 0);
#endif
}
#endif // HAVE_TDM

// ---------------- WMMA fragment helpers (CDNA5 16x16x32 bf16 layouts) -----
__device__ __forceinline__ v16bf load_a(const __bf16* base, int stride, int row0, int kbase) {
  int lane = threadIdx.x & 31;
  int half = lane >> 4;
  const __bf16* p = base + (row0 + (lane & 15)) * stride + kbase + half * 8;
  v8bf lo = *(const v8bf*)p;
  v8bf hi = *(const v8bf*)(p + 16);
  v16bf r;
#pragma unroll
  for (int i = 0; i < 8; ++i) { r[i] = lo[i]; r[i + 8] = hi[i]; }
  return r;
}
__device__ __forceinline__ v16bf load_b(const __bf16* base, int stride, int col0, int kbase) {
  int lane = threadIdx.x & 31;
  int half = lane >> 4;
  const __bf16* p = base + (col0 + (lane & 15)) * stride + kbase + half * 8;
  v8bf lo = *(const v8bf*)p;
  v8bf hi = *(const v8bf*)(p + 16);
  v16bf r;
#pragma unroll
  for (int i = 0; i < 8; ++i) { r[i] = lo[i]; r[i + 8] = hi[i]; }
  return r;
}
__device__ __forceinline__ v8f wmma_bf16(v16bf a, v16bf b, v8f c) {
  return __builtin_amdgcn_wmma_f32_16x16x32_bf16(false, a, false, b, (short)0, c, false, false);
}
__device__ __forceinline__ void store_rm(__bf16* dst, int stride, const float* bias,
                                         v8f acc, int mt, int nt) {
  int lane = threadIdx.x & 31;
  int hi = lane >> 4;
  int o = nt * 16 + (lane & 15);
  float bb = bias ? bias[o] : 0.f;
  int l0 = mt * 16 + hi * 8;
#pragma unroll
  for (int r = 0; r < 8; ++r) dst[(l0 + r) * stride + o] = f2bf(acc[r] + bb);
}
__device__ __forceinline__ void store_tr(__bf16* dst, int stride, const float* bias,
                                         v8f acc, int mt, int nt) {
  int lane = threadIdx.x & 31;
  int hi = lane >> 4;
  int o = nt * 16 + (lane & 15);
  float bb = bias ? bias[o] : 0.f;
  int l0 = mt * 16 + hi * 8;
  v8bf pk;
#pragma unroll
  for (int r = 0; r < 8; ++r) pk[r] = f2bf(acc[r] + bb);
  *(v8bf*)(dst + o * stride + l0) = pk;
}
__device__ __forceinline__ void store_f32(float* dst, int stride, v8f acc, int mt, int nt) {
  int lane = threadIdx.x & 31;
  int hi = lane >> 4;
  int col = nt * 16 + (lane & 15);
  int r0 = mt * 16 + hi * 8;
#pragma unroll
  for (int r = 0; r < 8; ++r) dst[(r0 + r) * stride + col] = acc[r];
}
__device__ __forceinline__ void store_tr_f32(float* dst, int stride, v8f acc, int mt, int nt) {
  int lane = threadIdx.x & 31;
  int hi = lane >> 4;
  int o = nt * 16 + (lane & 15);
  int l0 = mt * 16 + hi * 8;
  float4 u0, u1;
  u0.x = acc[0]; u0.y = acc[1]; u0.z = acc[2]; u0.w = acc[3];
  u1.x = acc[4]; u1.y = acc[5]; u1.z = acc[6]; u1.w = acc[7];
  *(float4*)(dst + o * stride + l0)     = u0;
  *(float4*)(dst + o * stride + l0 + 4) = u1;
}

__global__ void convert_weights(const float* __restrict__ Wq, const float* __restrict__ Wk,
                                const float* __restrict__ Wv, __bf16* __restrict__ o) {
  int i = blockIdx.x * 256 + threadIdx.x;
  if (i < CD * CD) {
    o[i]               = f2bf(Wq[i]);
    o[CD * CD + i]     = f2bf(Wk[i]);
    o[2 * CD * CD + i] = f2bf(Wv[i]);
  }
}

__global__ __launch_bounds__(256) void camixer_fused(
    const float* __restrict__ x,
    const __bf16* __restrict__ wbf,
    const float* __restrict__ bq, const float* __restrict__ bk, const float* __restrict__ bv,
    float* __restrict__ out) {
  extern __shared__ char smem[];
  __bf16* sW   = (__bf16*)(smem + OFF_W);
  __bf16* sWq  = sW;
  __bf16* sWk  = sW + 9216;
  __bf16* sWv  = sW + 18432;
  float*  stg0 = (float*)(smem + OFF_S0);
  float*  stg1 = (float*)(smem + OFF_S1);
  __bf16* sX   = (__bf16*)(smem + OFF_SX);
  __bf16* sQ   = (__bf16*)(smem + OFF_SQ);
  __bf16* sK   = (__bf16*)(smem + OFF_SK);
  __bf16* sVT  = (__bf16*)(smem + OFF_SVT);
  __bf16* sAg  = (__bf16*)(smem + OFF_SAG);
  __bf16* sA1  = (__bf16*)(smem + OFF_SA1);
  __bf16* sAVT = (__bf16*)(smem + OFF_SAVT);
  __bf16* sA2  = (__bf16*)(smem + OFF_SA2);
  float*  sL1  = (float*)(smem + OFF_SL1);
  float*  sL2  = (float*)(smem + OFF_SL2);
  float*  sFT  = (float*)(smem + OFF_SFT);

  const int tid  = threadIdx.x;
  const int lane = tid & 31;
  const int wave = tid >> 5;
  const int mt   = wave >> 1;
  const int nb0  = (wave & 1) * 3;

#if HAVE_TDM
  // ---- prologue: TDM weights (1D) + first x tile (3D, 8x8x96) -----------
  if (wave == 0) {
    {
      v4u g0 = tdm_g0(OFF_W, (unsigned long long)wbf);
      v8i g1 = tdm_g1(1, 27648u, 1u, 27648u, 0u, 0u, 27648ull, 27648ull);
      v4i g2 = tdm_g2(1u);
      v4i g3 = {0, 0, 0, 0};
      tdm_load(g0, g1, g2, g3);
    }
    {
      int win = blockIdx.x * WINS_PER_BLOCK;
      int b = win / 576, wy = (win / 24) % 24, wx = win % 24;
      unsigned long long ga = (unsigned long long)x +
          4ull * (unsigned)(b * CD * HW + wy * 8 * WWD + wx * 8);
      v4u g0 = tdm_g0(OFF_S0, ga);
      v8i g1 = tdm_g1(2, (unsigned)WWD, (unsigned)HH, 8u, 8u, (unsigned)CD,
                      (unsigned long long)WWD, (unsigned long long)HW);
      v4i g2 = tdm_g2((unsigned)CD);
      v4i g3 = {0, 0, 0, 0};
      tdm_load(g0, g1, g2, g3);
    }
  }
#else
  { // fallback: cooperative weight copy
    const v8bf* src = (const v8bf*)wbf;
    v8bf* dst = (v8bf*)sW;
    for (int i = tid; i < 3456; i += 256) dst[i] = src[i];
  }
#endif

  for (int it = 0; it < WINS_PER_BLOCK; ++it) {
    const int win = blockIdx.x * WINS_PER_BLOCK + it;
    const int b  = win / 576;
    const int wy = (win / 24) % 24;
    const int wx = win % 24;

#if HAVE_TDM
    // prefetch next tile, then wait for: current tile, weights, prior f-store
    if (wave == 0) {
      if (it + 1 < WINS_PER_BLOCK) {
        int wn = win + 1;
        int nb = wn / 576, ny = (wn / 24) % 24, nx = wn % 24;
        unsigned long long ga = (unsigned long long)x +
            4ull * (unsigned)(nb * CD * HW + ny * 8 * WWD + nx * 8);
        v4u g0 = tdm_g0(((it + 1) & 1) ? OFF_S1 : OFF_S0, ga);
        v8i g1 = tdm_g1(2, (unsigned)WWD, (unsigned)HH, 8u, 8u, (unsigned)CD,
                        (unsigned long long)WWD, (unsigned long long)HW);
        v4i g2 = tdm_g2((unsigned)CD);
        v4i g3 = {0, 0, 0, 0};
        tdm_load(g0, g1, g2, g3);
        __builtin_amdgcn_s_wait_tensorcnt(1);
      } else {
        __builtin_amdgcn_s_wait_tensorcnt(0);
      }
    }
    __syncthreads();
    { // stage f32 [c][l] -> sX bf16 [l][c]
      const float* stgb = (it & 1) ? stg1 : stg0;
      for (int e = tid; e < 6144; e += 256) {
        int c = e >> 6, l = e & 63;
        sX[l * CD + c] = f2bf(stgb[c * 64 + l]);
      }
    }
#else
    { // fallback: direct global load + convert
      const float* xb = x + b * CD * HW + (wy * 8) * WWD + wx * 8;
      for (int e = tid; e < 6144; e += 256) {
        int c = e >> 6, l = e & 63;
        int dh = l >> 3, dw = l & 7;
        sX[l * CD + c] = f2bf(xb[c * HW + dh * WWD + dw]);
      }
    }
#endif
    __syncthreads();

    // ---- q / k / v projections ----
    {
      v8f a0 = vzero(), a1 = vzero(), a2 = vzero();
#pragma unroll
      for (int ks = 0; ks < 3; ++ks) {
        v16bf a = load_a(sX, CD, mt * 16, ks * 32);
        a0 = wmma_bf16(a, load_b(sWq, CD, (nb0 + 0) * 16, ks * 32), a0);
        a1 = wmma_bf16(a, load_b(sWq, CD, (nb0 + 1) * 16, ks * 32), a1);
        a2 = wmma_bf16(a, load_b(sWq, CD, (nb0 + 2) * 16, ks * 32), a2);
      }
      store_rm(sQ, CD, bq, a0, mt, nb0 + 0);
      store_rm(sQ, CD, bq, a1, mt, nb0 + 1);
      store_rm(sQ, CD, bq, a2, mt, nb0 + 2);
    }
    {
      v8f a0 = vzero(), a1 = vzero(), a2 = vzero();
#pragma unroll
      for (int ks = 0; ks < 3; ++ks) {
        v16bf a = load_a(sX, CD, mt * 16, ks * 32);
        a0 = wmma_bf16(a, load_b(sWk, CD, (nb0 + 0) * 16, ks * 32), a0);
        a1 = wmma_bf16(a, load_b(sWk, CD, (nb0 + 1) * 16, ks * 32), a1);
        a2 = wmma_bf16(a, load_b(sWk, CD, (nb0 + 2) * 16, ks * 32), a2);
      }
      store_rm(sK, CD, bk, a0, mt, nb0 + 0);
      store_rm(sK, CD, bk, a1, mt, nb0 + 1);
      store_rm(sK, CD, bk, a2, mt, nb0 + 2);
    }
    {
      v8f a0 = vzero(), a1 = vzero(), a2 = vzero();
#pragma unroll
      for (int ks = 0; ks < 3; ++ks) {
        v16bf a = load_a(sX, CD, mt * 16, ks * 32);
        a0 = wmma_bf16(a, load_b(sWv, CD, (nb0 + 0) * 16, ks * 32), a0);
        a1 = wmma_bf16(a, load_b(sWv, CD, (nb0 + 1) * 16, ks * 32), a1);
        a2 = wmma_bf16(a, load_b(sWv, CD, (nb0 + 2) * 16, ks * 32), a2);
      }
      store_tr(sVT, 64, bv, a0, mt, nb0 + 0);
      store_tr(sVT, 64, bv, a1, mt, nb0 + 1);
      store_tr(sVT, 64, bv, a2, mt, nb0 + 2);
    }
    __syncthreads();

    // ---- agent pooling (adaptive avg pool 64 -> 7), exact f32 ----
    for (int e = tid; e < 7 * CD; e += 256) {
      int p = e / CD, o = e % CD;
      int s  = (p * 64) / 7;
      int en = ((p + 1) * 64 + 6) / 7;
      float sum = 0.f;
      for (int l = s; l < en; ++l) sum += bf2f(sQ[l * CD + o]);
      sAg[p * CD + o] = f2bf(sum / (float)(en - s));
    }
    for (int e = tid; e < 9 * CD; e += 256)
      sAg[(7 + e / CD) * CD + (e % CD)] = f2bf(0.f);
    __syncthreads();

    // ---- attn1 logits (agent @ k^T) | attn2 logits (q @ agent^T) ----
    if (wave < 4) {
      v8f acc = vzero();
#pragma unroll
      for (int ks = 0; ks < 3; ++ks)
        acc = wmma_bf16(load_a(sAg, CD, 0, ks * 32), load_b(sK, CD, wave * 16, ks * 32), acc);
      store_f32(sL1, 64, acc, 0, wave);
    } else {
      int m = wave - 4;
      v8f acc = vzero();
#pragma unroll
      for (int ks = 0; ks < 3; ++ks)
        acc = wmma_bf16(load_a(sQ, CD, m * 16, ks * 32), load_b(sAg, CD, 0, ks * 32), acc);
      store_f32(sL2, 16, acc, m, 0);
    }
    __syncthreads();

    // ---- softmaxes + zero padding ----
    for (int row = wave; row < 16; row += 8) {
      float v0 = sL1[row * 64 + lane];
      float v1 = sL1[row * 64 + 32 + lane];
      float mx = fmaxf(v0, v1);
      for (int off = 16; off; off >>= 1) mx = fmaxf(mx, __shfl_xor(mx, off));
      float e0 = __expf(v0 - mx), e1 = __expf(v1 - mx);
      float sm = e0 + e1;
      for (int off = 16; off; off >>= 1) sm += __shfl_xor(sm, off);
      float inv = 1.f / sm;
      sA1[row * 64 + lane]      = f2bf(e0 * inv);
      sA1[row * 64 + 32 + lane] = f2bf(e1 * inv);
    }
    if (tid < 64) {
      float ev[7];
      float mx = -1e30f;
      for (int p = 0; p < 7; ++p) mx = fmaxf(mx, sL2[tid * 16 + p]);
      float sm = 0.f;
      for (int p = 0; p < 7; ++p) { ev[p] = __expf(sL2[tid * 16 + p] - mx); sm += ev[p]; }
      float inv = 1.f / sm;
      for (int p = 0; p < 7; ++p)  sA2[tid * 32 + p] = f2bf(ev[p] * inv);
      for (int p = 7; p < 32; ++p) sA2[tid * 32 + p] = f2bf(0.f);
    }
    for (int e = tid; e < CD * 16; e += 256) {
      int o = e >> 4, p = 16 + (e & 15);
      sAVT[o * 32 + p] = f2bf(0.f);
    }
    __syncthreads();

    // ---- agent_v = attn1(16x64) @ v(64x96), stored transposed ----
    if (wave < 6) {
      v8f acc = vzero();
#pragma unroll
      for (int ks = 0; ks < 2; ++ks)
        acc = wmma_bf16(load_a(sA1, 64, 0, ks * 32), load_b(sVT, 64, wave * 16, ks * 32), acc);
      store_tr(sAVT, 32, nullptr, acc, 0, wave);
    }
    __syncthreads();

    // ---- f = attn2(64x32, K-padded) @ agent_v(32x96) -> f^T [96][64] ----
    {
      v16bf a = load_a(sA2, 32, mt * 16, 0);
#pragma unroll
      for (int j = 0; j < 3; ++j) {
        v8f acc = vzero();
        acc = wmma_bf16(a, load_b(sAVT, 32, (nb0 + j) * 16, 0), acc);
        store_tr_f32(sFT, 64, acc, mt, nb0 + j);
      }
    }
    __syncthreads();

    // ---- output: TDM store of f^T tile (same 3D descriptor vs out) ----
#if HAVE_TDM
    if (wave == 0) {
      unsigned long long ga = (unsigned long long)out +
          4ull * (unsigned)(b * CD * HW + wy * 8 * WWD + wx * 8);
      v4u g0 = tdm_g0(OFF_SFT, ga);
      v8i g1 = tdm_g1(2, (unsigned)WWD, (unsigned)HH, 8u, 8u, (unsigned)CD,
                      (unsigned long long)WWD, (unsigned long long)HW);
      v4i g2 = tdm_g2((unsigned)CD);
      v4i g3 = {0, 0, 0, 0};
      tdm_store(g0, g1, g2, g3);
      // completion enforced by next iteration's s_wait_tensorcnt (in-order)
      // and by the implicit wait-idle at S_ENDPGM on the last iteration.
    }
#else
    {
      float* ob = out + b * CD * HW + (wy * 8) * WWD + wx * 8;
      for (int e = tid; e < CD * 8; e += 256) {
        int o = e >> 3, dh = e & 7;
        const float* fp = sFT + o * 64 + dh * 8;
        float4 u0 = *(const float4*)fp;
        float4 u1 = *(const float4*)(fp + 4);
        float* gp = ob + o * HW + dh * WWD;
        *(float4*)gp       = u0;
        *(float4*)(gp + 4) = u1;
      }
    }
#endif
    __syncthreads();
  }
}

extern "C" void kernel_launch(void* const* d_in, const int* in_sizes, int n_in,
                              void* d_out, int out_size, void* d_ws, size_t ws_size,
                              hipStream_t stream) {
  (void)in_sizes; (void)n_in; (void)out_size; (void)ws_size;
  const float* x  = (const float*)d_in[0];
  const float* Wv = (const float*)d_in[1];
  const float* bv = (const float*)d_in[2];
  const float* Wq = (const float*)d_in[3];
  const float* bq = (const float*)d_in[4];
  const float* Wk = (const float*)d_in[5];
  const float* bk = (const float*)d_in[6];
  float* out = (float*)d_out;
  __bf16* wbf = (__bf16*)d_ws;   // 27648 bf16 = 55296 bytes of scratch

  convert_weights<<<(CD * CD + 255) / 256, 256, 0, stream>>>(Wq, Wk, Wv, wbf);

  hipFuncSetAttribute((const void*)camixer_fused,
                      hipFuncAttributeMaxDynamicSharedMemorySize, (int)SMEM_BYTES);
  camixer_fused<<<NBLK, 256, SMEM_BYTES, stream>>>(x, wbf, bq, bk, bv, out);
}
